// MultiHeadAttention_33878702031101
// MI455X (gfx1250) — matrix operand order repaired and checked by reading, not compiled, not run on hardware
//
#include <hip/hip_runtime.h>
#include <hip/hip_bf16.h>
#include <stdint.h>

// ---------------------------------------------------------------------------
// MHA block for MI455X (gfx1250, wave32, WMMA).
// All matmuls use v_wmma_f32_16x16x32_bf16 (fp32 accumulate).
// ---------------------------------------------------------------------------

typedef __attribute__((ext_vector_type(16))) __bf16 v16bf;
typedef __attribute__((ext_vector_type(8)))  float  v8f;

#define N_HEADS 16
#define D_MODEL 1024
#define D_K     64
#define SEQ     2048
#define BATCH   2
#define ROWS    (BATCH * SEQ)        // 4096
#define LN_EPS  1e-6f
#define NEG_INF_F (-1e9f)

__device__ __forceinline__ unsigned short f32_to_bf16(float f) {
  unsigned int u = __float_as_uint(f);
  u += 0x7FFFu + ((u >> 16) & 1u);   // round-to-nearest-even
  return (unsigned short)(u >> 16);
}

union Frag16 {            // 16 bf16 = 8 VGPRs: one WMMA A or B operand
  v16bf v;
  uint4 u[2];
};

// ---------------------------------------------------------------------------
// Generic row-major GEMM:  C[M,N] = A[M,K] @ B[K,N]  (+ optional residual)
// Block tile 128x128, 8 waves, each wave 64x32 (4x2 WMMA tiles), K-step 32.
// fp32 inputs converted to bf16 on the fly while staging into LDS.
// global_prefetch of the next K tile hides HBM/L2 latency.
// ---------------------------------------------------------------------------
__global__ __launch_bounds__(256)
void gemm_rowmajor_bf16(const float* __restrict__ A, const float* __restrict__ Bw,
                        float* __restrict__ C, const float* __restrict__ resid,
                        int M, int N, int K)
{
  (void)M;
  __shared__ __align__(16) unsigned short ldsA[128 * 32];   // [m][k]
  __shared__ __align__(16) unsigned short ldsB[128 * 32];   // [n][k]  (B transposed)

  const int tid  = threadIdx.x;
  const int wave = tid >> 5, lane = tid & 31;
  const int r    = lane & 15, hilane = lane >> 4;
  const int waveM = wave >> 2;     // 0..1  -> 64 rows each
  const int waveN = wave & 3;      // 0..3  -> 32 cols each
  const int bM = blockIdx.y * 128;
  const int bN = blockIdx.x * 128;

  v8f zero = {};
  v8f acc[4][2];
  #pragma unroll
  for (int i = 0; i < 4; ++i)
    #pragma unroll
    for (int j = 0; j < 2; ++j) acc[i][j] = zero;

  for (int k0 = 0; k0 < K; k0 += 32) {
    const bool more = (k0 + 32) < K;
    // ---- stage A tile: 128x32 fp32 -> bf16 (row-major) ----
    #pragma unroll
    for (int it = 0; it < 4; ++it) {
      int g   = tid + it * 256;            // group of 4 elements
      int row = g >> 3;                    // 8 groups per 32-wide row
      int cg  = (g & 7) << 2;
      const float* src = &A[(size_t)(bM + row) * K + k0 + cg];
      float4 f = *(const float4*)src;
      if (more) __builtin_prefetch(src + 32, 0, 3);   // next K tile -> global_prefetch_b8
      unsigned int lo = (unsigned int)f32_to_bf16(f.x) | ((unsigned int)f32_to_bf16(f.y) << 16);
      unsigned int hw = (unsigned int)f32_to_bf16(f.z) | ((unsigned int)f32_to_bf16(f.w) << 16);
      *(uint2*)&ldsA[row * 32 + cg] = make_uint2(lo, hw);
    }
    // ---- stage B tile: 32x128 fp32 -> bf16 transposed into [n][k] ----
    #pragma unroll
    for (int it = 0; it < 4; ++it) {
      int g   = tid + it * 256;
      int row = g >> 5;                    // k index 0..31
      int cg  = (g & 31) << 2;             // n offset
      const float* src = &Bw[(size_t)(k0 + row) * N + bN + cg];
      float4 f = *(const float4*)src;
      if (more) __builtin_prefetch(src + (size_t)32 * N, 0, 3);
      ldsB[(cg + 0) * 32 + row] = f32_to_bf16(f.x);
      ldsB[(cg + 1) * 32 + row] = f32_to_bf16(f.y);
      ldsB[(cg + 2) * 32 + row] = f32_to_bf16(f.z);
      ldsB[(cg + 3) * 32 + row] = f32_to_bf16(f.w);
    }
    __syncthreads();

    Frag16 bf[2];
    #pragma unroll
    for (int j = 0; j < 2; ++j) {
      int n = waveN * 32 + j * 16 + r;
      bf[j].u[0] = *(const uint4*)&ldsB[n * 32 + hilane * 8];
      bf[j].u[1] = *(const uint4*)&ldsB[n * 32 + 16 + hilane * 8];
    }
    #pragma unroll
    for (int i = 0; i < 4; ++i) {
      int m = waveM * 64 + i * 16 + r;
      Frag16 af;
      af.u[0] = *(const uint4*)&ldsA[m * 32 + hilane * 8];        // K 0..7 / 8..15
      af.u[1] = *(const uint4*)&ldsA[m * 32 + 16 + hilane * 8];   // K 16..23 / 24..31
      #pragma unroll
      for (int j = 0; j < 2; ++j)
        acc[i][j] = __builtin_amdgcn_wmma_f32_16x16x32_bf16(
            false, af.v, false, bf[j].v, (short)0, acc[i][j], false, false);
    }
    __syncthreads();
  }

  // Epilogue: residual branch hoisted (uniform) out of the store loops.
  if (resid) {
    #pragma unroll
    for (int i = 0; i < 4; ++i)
      #pragma unroll
      for (int j = 0; j < 2; ++j) {
        int mb = bM + waveM * 64 + i * 16 + hilane * 8;
        int n  = bN + waveN * 32 + j * 16 + r;
        #pragma unroll
        for (int e = 0; e < 8; ++e) {
          size_t idx = (size_t)(mb + e) * N + n;
          C[idx] = acc[i][j][e] + resid[idx];
        }
      }
  } else {
    #pragma unroll
    for (int i = 0; i < 4; ++i)
      #pragma unroll
      for (int j = 0; j < 2; ++j) {
        int mb = bM + waveM * 64 + i * 16 + hilane * 8;
        int n  = bN + waveN * 32 + j * 16 + r;
        #pragma unroll
        for (int e = 0; e < 8; ++e)
          C[(size_t)(mb + e) * N + n] = acc[i][j][e];
      }
  }
}

// ---------------------------------------------------------------------------
// Scores: attn_raw[b,h,q,key] = (Q[b,q,h,:] . K[b,key,h,:]) / 8, masked.
// K=64 (2 K-steps). B-tile staging is a straight row copy of K (QK^T!).
// Epilogue bounces the 128x128 tile through LDS (two 128x64 halves) so the
// dominant 537MB attn write goes out as fully coalesced float4 stores, with
// the mask bytes read coalesced as uchar4 in the same pass.
// ---------------------------------------------------------------------------
__global__ __launch_bounds__(256)
void attn_scores(const float* __restrict__ Qb, const float* __restrict__ Kb,
                 const unsigned char* __restrict__ mask, float* __restrict__ attn)
{
  __shared__ __align__(16) unsigned short ldsA[128 * 32];   // [q][kd]
  __shared__ __align__(16) unsigned short ldsB[128 * 32];   // [key][kd]
  __shared__ __align__(16) float          ldsC[128 * 64];   // C-tile bounce (half)
  const int tid  = threadIdx.x;
  const int wave = tid >> 5, lane = tid & 31;
  const int r = lane & 15, hilane = lane >> 4;
  const int waveM = wave >> 2, waveN = wave & 3;
  const int bh = blockIdx.z, b = bh >> 4, h = bh & 15;
  const int bM = blockIdx.y * 128, bN = blockIdx.x * 128;

  const float* Qh = Qb + (size_t)b * SEQ * D_MODEL + h * D_K;
  const float* Kh = Kb + (size_t)b * SEQ * D_MODEL + h * D_K;

  v8f zero = {};
  v8f acc[4][2];
  #pragma unroll
  for (int i = 0; i < 4; ++i)
    #pragma unroll
    for (int j = 0; j < 2; ++j) acc[i][j] = zero;

  #pragma unroll
  for (int k0 = 0; k0 < D_K; k0 += 32) {
    #pragma unroll
    for (int it = 0; it < 4; ++it) {
      int g = tid + it * 256;
      int row = g >> 3, cg = (g & 7) << 2;
      float4 f = *(const float4*)&Qh[(size_t)(bM + row) * D_MODEL + k0 + cg];
      unsigned int lo = (unsigned int)f32_to_bf16(f.x) | ((unsigned int)f32_to_bf16(f.y) << 16);
      unsigned int hw = (unsigned int)f32_to_bf16(f.z) | ((unsigned int)f32_to_bf16(f.w) << 16);
      *(uint2*)&ldsA[row * 32 + cg] = make_uint2(lo, hw);
    }
    #pragma unroll
    for (int it = 0; it < 4; ++it) {
      int g = tid + it * 256;
      int row = g >> 3, cg = (g & 7) << 2;
      float4 f = *(const float4*)&Kh[(size_t)(bN + row) * D_MODEL + k0 + cg];
      unsigned int lo = (unsigned int)f32_to_bf16(f.x) | ((unsigned int)f32_to_bf16(f.y) << 16);
      unsigned int hw = (unsigned int)f32_to_bf16(f.z) | ((unsigned int)f32_to_bf16(f.w) << 16);
      *(uint2*)&ldsB[row * 32 + cg] = make_uint2(lo, hw);
    }
    __syncthreads();

    Frag16 bf[2];
    #pragma unroll
    for (int j = 0; j < 2; ++j) {
      int n = waveN * 32 + j * 16 + r;
      bf[j].u[0] = *(const uint4*)&ldsB[n * 32 + hilane * 8];
      bf[j].u[1] = *(const uint4*)&ldsB[n * 32 + 16 + hilane * 8];
    }
    #pragma unroll
    for (int i = 0; i < 4; ++i) {
      int m = waveM * 64 + i * 16 + r;
      Frag16 af;
      af.u[0] = *(const uint4*)&ldsA[m * 32 + hilane * 8];
      af.u[1] = *(const uint4*)&ldsA[m * 32 + 16 + hilane * 8];
      #pragma unroll
      for (int j = 0; j < 2; ++j)
        acc[i][j] = __builtin_amdgcn_wmma_f32_16x16x32_bf16(
            false, af.v, false, bf[j].v, (short)0, acc[i][j], false, false);
    }
    __syncthreads();
  }

  float* Ap = attn + (size_t)bh * SEQ * SEQ;
  // Two 128x64 halves through LDS -> coalesced stores.
  #pragma unroll
  for (int half = 0; half < 2; ++half) {
    __syncthreads();
    if ((waveN >> 1) == half) {
      #pragma unroll
      for (int i = 0; i < 4; ++i)
        #pragma unroll
        for (int j = 0; j < 2; ++j) {
          int mb = waveM * 64 + i * 16 + hilane * 8;
          int n  = (waveN & 1) * 32 + j * 16 + r;
          #pragma unroll
          for (int e = 0; e < 8; ++e)
            ldsC[(mb + e) * 64 + n] = acc[i][j][e] * 0.125f;
        }
    }
    __syncthreads();
    #pragma unroll
    for (int it = 0; it < 8; ++it) {
      int g   = tid + it * 256;
      int row = g >> 4;                 // 16 float4 groups per 64-wide row
      int cg  = (g & 15) << 2;
      int qrow = bM + row;
      int key  = bN + half * 64 + cg;
      float4 vals = *(const float4*)&ldsC[row * 64 + cg];
      uchar4 mv = *(const uchar4*)&mask[((size_t)b * SEQ + qrow) * SEQ + key];
      float4 o;
      o.x = mv.x ? vals.x : NEG_INF_F;
      o.y = mv.y ? vals.y : NEG_INF_F;
      o.z = mv.z ? vals.z : NEG_INF_F;
      o.w = mv.w ? vals.w : NEG_INF_F;
      *(float4*)&Ap[(size_t)qrow * SEQ + key] = o;
    }
  }
}

// ---------------------------------------------------------------------------
// In-place softmax over 2048-wide rows of attn. One block per (b,h,q) row.
// ---------------------------------------------------------------------------
__global__ __launch_bounds__(256)
void softmax_rows(float* __restrict__ attn)
{
  __shared__ float red[256];
  const int tid = threadIdx.x;
  float* p = attn + (size_t)blockIdx.x * SEQ;

  float loc[8];
  float mx = -3.4e38f;
  #pragma unroll
  for (int i = 0; i < 8; ++i) { loc[i] = p[tid + i * 256]; mx = fmaxf(mx, loc[i]); }
  red[tid] = mx; __syncthreads();
  for (int s = 128; s > 0; s >>= 1) {
    if (tid < s) red[tid] = fmaxf(red[tid], red[tid + s]);
    __syncthreads();
  }
  mx = red[0]; __syncthreads();

  float sum = 0.f;
  #pragma unroll
  for (int i = 0; i < 8; ++i) { loc[i] = __expf(loc[i] - mx); sum += loc[i]; }
  red[tid] = sum; __syncthreads();
  for (int s = 128; s > 0; s >>= 1) {
    if (tid < s) red[tid] += red[tid + s];
    __syncthreads();
  }
  float inv = 1.0f / red[0];
  #pragma unroll
  for (int i = 0; i < 8; ++i) p[tid + i * 256] = loc[i] * inv;
}

// ---------------------------------------------------------------------------
// PV: cat[b,q,h*64+d] = sum_key attn[b,h,q,key] * V[b,key,h,d]
// Tile 128(q) x 64(d), waves 4x2 (each 32x32 = 2x2 WMMA tiles), K loop 2048.
// ---------------------------------------------------------------------------
__global__ __launch_bounds__(256)
void attn_value(const float* __restrict__ attn, const float* __restrict__ Vb,
                float* __restrict__ cat)
{
  __shared__ __align__(16) unsigned short ldsA[128 * 32];   // [q][key]
  __shared__ __align__(16) unsigned short ldsB[64 * 32];    // [d][key] (transposed V)
  const int tid  = threadIdx.x;
  const int wave = tid >> 5, lane = tid & 31;
  const int r = lane & 15, hilane = lane >> 4;
  const int waveM = wave >> 1;   // 0..3 -> 32 rows each
  const int waveN = wave & 1;    // 0..1 -> 32 cols each
  const int bh = blockIdx.y, b = bh >> 4, h = bh & 15;
  const int bM = blockIdx.x * 128;

  const float* Ap = attn + (size_t)bh * SEQ * SEQ;
  const float* Vh = Vb + (size_t)b * SEQ * D_MODEL + h * D_K;

  v8f zero = {};
  v8f acc[2][2];
  #pragma unroll
  for (int i = 0; i < 2; ++i)
    #pragma unroll
    for (int j = 0; j < 2; ++j) acc[i][j] = zero;

  for (int k0 = 0; k0 < SEQ; k0 += 32) {
    const bool more = (k0 + 32) < SEQ;
    #pragma unroll
    for (int it = 0; it < 4; ++it) {
      int g = tid + it * 256;
      int row = g >> 3, cg = (g & 7) << 2;
      const float* src = &Ap[(size_t)(bM + row) * SEQ + k0 + cg];
      float4 f = *(const float4*)src;
      if (more) __builtin_prefetch(src + 32, 0, 3);
      unsigned int lo = (unsigned int)f32_to_bf16(f.x) | ((unsigned int)f32_to_bf16(f.y) << 16);
      unsigned int hw = (unsigned int)f32_to_bf16(f.z) | ((unsigned int)f32_to_bf16(f.w) << 16);
      *(uint2*)&ldsA[row * 32 + cg] = make_uint2(lo, hw);
    }
    #pragma unroll
    for (int it = 0; it < 2; ++it) {
      int g = tid + it * 256;
      int row = g >> 4;                    // key index 0..31
      int cg  = (g & 15) << 2;             // d offset
      const float* src = &Vh[(size_t)(k0 + row) * D_MODEL + cg];
      float4 f = *(const float4*)src;
      if (more) __builtin_prefetch(src + (size_t)32 * D_MODEL, 0, 3);
      ldsB[(cg + 0) * 32 + row] = f32_to_bf16(f.x);
      ldsB[(cg + 1) * 32 + row] = f32_to_bf16(f.y);
      ldsB[(cg + 2) * 32 + row] = f32_to_bf16(f.z);
      ldsB[(cg + 3) * 32 + row] = f32_to_bf16(f.w);
    }
    __syncthreads();

    Frag16 bf[2];
    #pragma unroll
    for (int j = 0; j < 2; ++j) {
      int n = waveN * 32 + j * 16 + r;
      bf[j].u[0] = *(const uint4*)&ldsB[n * 32 + hilane * 8];
      bf[j].u[1] = *(const uint4*)&ldsB[n * 32 + 16 + hilane * 8];
    }
    #pragma unroll
    for (int i = 0; i < 2; ++i) {
      int m = waveM * 32 + i * 16 + r;
      Frag16 af;
      af.u[0] = *(const uint4*)&ldsA[m * 32 + hilane * 8];
      af.u[1] = *(const uint4*)&ldsA[m * 32 + 16 + hilane * 8];
      #pragma unroll
      for (int j = 0; j < 2; ++j)
        acc[i][j] = __builtin_amdgcn_wmma_f32_16x16x32_bf16(
            false, af.v, false, bf[j].v, (short)0, acc[i][j], false, false);
    }
    __syncthreads();
  }

  #pragma unroll
  for (int i = 0; i < 2; ++i)
    #pragma unroll
    for (int j = 0; j < 2; ++j) {
      int qb = bM + waveM * 32 + i * 16 + hilane * 8;
      int n  = waveN * 32 + j * 16 + r;
      #pragma unroll
      for (int e = 0; e < 8; ++e)
        cat[((size_t)b * SEQ + qb + e) * D_MODEL + h * D_K + n] = acc[i][j][e];
    }
}

// ---------------------------------------------------------------------------
// LayerNorm over 1024-wide rows. One block per row.
// ---------------------------------------------------------------------------
__global__ __launch_bounds__(256)
void layernorm_rows(const float* __restrict__ x, const float* __restrict__ gamma,
                    const float* __restrict__ beta, float* __restrict__ out)
{
  __shared__ float red[256];
  const int tid = threadIdx.x;
  const float* px = x   + (size_t)blockIdx.x * D_MODEL;
  float*       po = out + (size_t)blockIdx.x * D_MODEL;

  float loc[4];
  float s = 0.f;
  #pragma unroll
  for (int i = 0; i < 4; ++i) { loc[i] = px[tid + i * 256]; s += loc[i]; }
  red[tid] = s; __syncthreads();
  for (int t = 128; t > 0; t >>= 1) {
    if (tid < t) red[tid] += red[tid + t];
    __syncthreads();
  }
  float mu = red[0] * (1.0f / D_MODEL);
  __syncthreads();

  float vs = 0.f;
  #pragma unroll
  for (int i = 0; i < 4; ++i) { float d = loc[i] - mu; vs += d * d; }
  red[tid] = vs; __syncthreads();
  for (int t = 128; t > 0; t >>= 1) {
    if (tid < t) red[tid] += red[tid + t];
    __syncthreads();
  }
  float inv = rsqrtf(red[0] * (1.0f / D_MODEL) + LN_EPS);

  #pragma unroll
  for (int i = 0; i < 4; ++i) {
    int c = tid + i * 256;
    po[c] = (loc[i] - mu) * inv * gamma[c] + beta[c];
  }
}

// ---------------------------------------------------------------------------
extern "C" void kernel_launch(void* const* d_in, const int* in_sizes, int n_in,
                              void* d_out, int out_size, void* d_ws, size_t ws_size,
                              hipStream_t stream)
{
  (void)in_sizes; (void)n_in; (void)out_size; (void)ws_size;

  const float*         q     = (const float*)d_in[0];
  const float*         k     = (const float*)d_in[1];
  const float*         v     = (const float*)d_in[2];
  const unsigned char* mask  = (const unsigned char*)d_in[3];
  const float*         Wq    = (const float*)d_in[4];
  const float*         Wk    = (const float*)d_in[5];
  const float*         Wv    = (const float*)d_in[6];
  const float*         Wo    = (const float*)d_in[7];
  const float*         gamma = (const float*)d_in[8];
  const float*         beta  = (const float*)d_in[9];

  float* out  = (float*)d_out;                       // [B,S,D_MODEL]
  float* attn = out + (size_t)ROWS * D_MODEL;        // [B,H,S,S]

  // workspace: 5 x 16 MB fp32 segments (Q, K, V, attn@V cat, pre-LN)
  float* ws  = (float*)d_ws;
  const size_t SEG = (size_t)ROWS * D_MODEL;
  float* Qb  = ws;
  float* Kb  = ws + SEG;
  float* Vb  = ws + 2 * SEG;
  float* Cat = ws + 3 * SEG;
  float* Pre = ws + 4 * SEG;

  dim3 blk(256);
  dim3 gProj(D_MODEL / 128, ROWS / 128);             // 8 x 32

  gemm_rowmajor_bf16<<<gProj, blk, 0, stream>>>(q, Wq, Qb, nullptr, ROWS, D_MODEL, D_MODEL);
  gemm_rowmajor_bf16<<<gProj, blk, 0, stream>>>(k, Wk, Kb, nullptr, ROWS, D_MODEL, D_MODEL);
  gemm_rowmajor_bf16<<<gProj, blk, 0, stream>>>(v, Wv, Vb, nullptr, ROWS, D_MODEL, D_MODEL);

  attn_scores<<<dim3(SEQ / 128, SEQ / 128, BATCH * N_HEADS), blk, 0, stream>>>(Qb, Kb, mask, attn);
  softmax_rows<<<dim3(BATCH * N_HEADS * SEQ), blk, 0, stream>>>(attn);
  attn_value<<<dim3(SEQ / 128, BATCH * N_HEADS), blk, 0, stream>>>(attn, Vb, Cat);

  gemm_rowmajor_bf16<<<gProj, blk, 0, stream>>>(Cat, Wo, Pre, q, ROWS, D_MODEL, D_MODEL);
  layernorm_rows<<<dim3(ROWS), blk, 0, stream>>>(Pre, gamma, beta, out);
}
